// ParticleNet_2542620639810
// MI455X (gfx1250) — compile-verified, matrix operand order checked by reading, not run on hardware
//
#include <hip/hip_runtime.h>
#include <hip/hip_bf16.h>
#include <math.h>

typedef __attribute__((ext_vector_type(16))) _Float16 v16h;
typedef __attribute__((ext_vector_type(8)))  _Float16 v8h;
typedef __attribute__((ext_vector_type(8)))  float    v8f;

#define BJ   128   // jets
#define NP   128   // particles per jet
#define KNN  16
#define BN_EPS 1e-5f

// ---------------------------------------------------------------------------
// Fragment layout (wave32 WMMA f32_16x16x32_f16)
// A frag (16x32 f16): lane L holds row m=L%16, half=L/16; element e maps to
//   k = (e>>3)*16 + half*8 + (e&7).  Inverse: half=(k&15)>>3,
//   e=((k>>4)<<3)|(k&7), lane=half*16+m.  So per (lane, kTile) the 16 halves
//   are two contiguous 8-half runs of the row: k0+half*8+{0..7}, +16.
// B frag (32x16 f16): same k-map, lane L holds col n=L%16.
// D frag (16x16 f32, v8f): lane L, vgpr r -> m=(L>>4)*8+r, n=L&15.
// Storage: frag[tileIdx][lane][16 halves] -> 32B contiguous per lane.
// ---------------------------------------------------------------------------

__device__ inline v16h join8(v8h lo, v8h hi) {
  v16h a;
#pragma unroll
  for (int e = 0; e < 8; ++e) { a[e] = lo[e]; a[e + 8] = hi[e]; }
  return a;
}

// ============================ input batchnorm =============================
__global__ void input_bn_kernel(const float* __restrict__ x,
                                const float* __restrict__ beta,
                                const float* __restrict__ gamma,
                                const float* __restrict__ mean,
                                const float* __restrict__ var,
                                float* __restrict__ f32, _Float16* __restrict__ f16) {
  int t = blockIdx.x * blockDim.x + threadIdx.x;
  if (t >= BJ * NP * 7) return;
  int c = t % 7;
  float s = gamma[c] * rsqrtf(var[c] + BN_EPS);
  float v = (x[t] - mean[c]) * s + beta[c];
  f32[t] = v;
  f16[t] = (_Float16)v;
}

// ============================ BN fold =====================================
__global__ void bnfold_kernel(const float* __restrict__ beta,
                              const float* __restrict__ gamma,
                              const float* __restrict__ mean,
                              const float* __restrict__ var,
                              float* __restrict__ scale, float* __restrict__ bias, int C) {
  int i = blockIdx.x * blockDim.x + threadIdx.x;
  if (i >= C) return;
  float s = gamma[i] * rsqrtf(var[i] + BN_EPS);
  scale[i] = s;
  bias[i]  = beta[i] - mean[i] * s;
}

// ================== weight f32 [K,Nc] -> f16 B-fragment layout ============
__global__ void w2frag_kernel(const float* __restrict__ src, _Float16* __restrict__ dst,
                              int K, int Kpad, int Nc) {
  int t = blockIdx.x * blockDim.x + threadIdx.x;
  if (t >= Kpad * Nc) return;
  int k = t / Nc, n = t % Nc;
  float v = (k < K) ? src[k * Nc + n] : 0.0f;
  int kTile = k >> 5, kk = k & 31;
  int half = (kk & 15) >> 3;
  int e    = ((kk >> 4) << 3) | (kk & 7);
  int nTile = n >> 4;
  int lane  = half * 16 + (n & 15);
  dst[(((size_t)kTile * (Nc / 16) + nTile) * 32 + lane) * 16 + e] = (_Float16)v;
}

// ============================ kNN per jet =================================
__global__ void __launch_bounds__(128)
knn_kernel(const float* __restrict__ pts, int D, int* __restrict__ idx) {
  extern __shared__ char smem[];
  float* dist  = (float*)smem;            // [128][132] padded rows
  float* tile  = dist + 128 * 132;        // [128][33]
  float* norms = tile + 128 * 33;         // [128]
  const int i = threadIdx.x;
  const int b = blockIdx.x;
  const float* P = pts + (size_t)b * NP * D;

  float d2 = 0.f;
  for (int c = 0; c < D; ++c) { float v = P[i * D + c]; d2 += v * v; }
  norms[i] = d2;
  __syncthreads();
  for (int j = 0; j < NP; ++j) dist[i * 132 + j] = d2 + norms[j];

  for (int c0 = 0; c0 < D; c0 += 32) {
    int cw = (D - c0) < 32 ? (D - c0) : 32;
    __syncthreads();
    for (int t = i; t < NP * cw; t += NP) {
      int r = t / cw, c = t % cw;
      tile[r * 33 + c] = P[r * D + c0 + c];
    }
    __syncthreads();
    float my[32];
    for (int c = 0; c < cw; ++c) my[c] = tile[i * 33 + c];
    for (int j = 0; j < NP; ++j) {
      float dot = 0.f;
      for (int c = 0; c < cw; ++c) dot += my[c] * tile[j * 33 + c];
      dist[i * 132 + j] -= 2.f * dot;
    }
  }
  __syncthreads();
  dist[i * 132 + i] = 1e30f;   // exclude self

  for (int t = 0; t < KNN; ++t) {
    float best = 1e29f; int bj = 0;
    for (int j = 0; j < NP; ++j) {
      float v = dist[i * 132 + j];
      if (v < best) { best = v; bj = j; }
    }
    idx[((size_t)b * NP + i) * KNN + t] = bj;
    dist[i * 132 + bj] = 1e30f;
  }
}

// ================= skip path: BN(fts16 @ Wskip)  (WMMA) ===================
// Templated so the A-fragment gather is branch-free; for D%8==0 each fragment
// is two contiguous 16B runs -> two global_load_b128 per kt.
template <int D, int KP, int COUT>
__global__ void __launch_bounds__(256)
skip_gemm_kernel(const _Float16* __restrict__ A16,
                 const _Float16* __restrict__ wfrag,
                 const float* __restrict__ scale, const float* __restrict__ bias,
                 float* __restrict__ out) {
  const int lane = threadIdx.x & 31, wave = threadIdx.x >> 5;
  constexpr int M = BJ * NP;              // 16384
  constexpr int rowTiles = M / 16;        // 1024
  int t = blockIdx.x * 8 + wave;
  if (t >= rowTiles * (COUT / 16)) return;
  int rowTile = t % rowTiles, colTile = t / rowTiles;

  const int m0 = rowTile * 16 + (lane & 15);
  const int halfsel = lane >> 4;
  const _Float16* rowP = A16 + (size_t)m0 * D;

  v8f acc = {};
#pragma unroll
  for (int kt = 0; kt < KP / 32; ++kt) {
    v16h a;
    if constexpr (D % 8 == 0) {
      const v8h lo = *(const v8h*)(rowP + kt * 32 + halfsel * 8);
      const v8h hi = *(const v8h*)(rowP + kt * 32 + 16 + halfsel * 8);
      a = join8(lo, hi);
    } else {
#pragma unroll
      for (int e = 0; e < 16; ++e) {
        int k = kt * 32 + ((e >> 3) << 4) + halfsel * 8 + (e & 7);
        a[e] = (k < D) ? rowP[k] : (_Float16)0.f;
      }
    }
    const v16h bf = *(const v16h*)(wfrag + (((size_t)kt * (COUT / 16) + colTile) * 32 + lane) * 16);
    acc = __builtin_amdgcn_wmma_f32_16x16x32_f16(false, a, false, bf, (short)0, acc, false, false);
  }
  int n = lane & 15, col = colTile * 16 + n;
  float s = scale[col], bb = bias[col];
#pragma unroll
  for (int r = 0; r < 8; ++r) {
    int m = rowTile * 16 + ((lane >> 4) << 3) + r;
    out[(size_t)m * COUT + col] = acc[r] * s + bb;   // no ReLU on skip branch
  }
}

// ============== fused EdgeConv: gather -> 3x(GEMM+BN+ReLU) -> kmax ========
template <int KP, int COUT>
__device__ inline void wg_gemm(const _Float16* srcFrag, _Float16* dstFrag,
                               const _Float16* __restrict__ wFrag,
                               const float* __restrict__ scale,
                               const float* __restrict__ bias,
                               int lane, int wave) {
  constexpr int KT  = KP / 32;     // k-tiles of src
  constexpr int CT  = COUT / 16;   // col-tiles of dst (even for all configs)
  constexpr int KTD = COUT / 32;   // k-tiles of dst (A-frag layout for next GEMM)
  const int rowTile = wave;        // 8 waves <-> 8 row tiles
  __builtin_prefetch(wFrag, 0, 1);

  v16h a[KT];
#pragma unroll
  for (int kt = 0; kt < KT; ++kt)
    a[kt] = *(const v16h*)(srcFrag + (((size_t)rowTile * KT + kt) * 32 + lane) * 16);

  const int n     = lane & 15;
  const int half  = n >> 3;
  const int mBase = ((lane >> 4) << 3);

  // two independent accumulator chains per iteration: fills WMMA RAW hazard
  // slots with the other chain's wmma + b128 loads.
  for (int ct = 0; ct < CT; ct += 2) {
    v8f acc0 = {}, acc1 = {};
#pragma unroll
    for (int kt = 0; kt < KT; ++kt) {
      const v16h b0 = *(const v16h*)(wFrag + (((size_t)kt * CT + ct)     * 32 + lane) * 16);
      const v16h b1 = *(const v16h*)(wFrag + (((size_t)kt * CT + ct + 1) * 32 + lane) * 16);
      acc0 = __builtin_amdgcn_wmma_f32_16x16x32_f16(false, a[kt], false, b0, (short)0, acc0, false, false);
      acc1 = __builtin_amdgcn_wmma_f32_16x16x32_f16(false, a[kt], false, b1, (short)0, acc1, false, false);
    }
#pragma unroll
    for (int which = 0; which < 2; ++which) {
      const v8f& acc  = which ? acc1 : acc0;
      const int col   = (ct + which) * 16 + n;
      const float s   = scale[col];
      const float bb  = bias[col];
      const int kTileD = col >> 5;
      const int e      = (((col & 31) >> 4) << 3) | (n & 7);
#pragma unroll
      for (int r = 0; r < 8; ++r) {
        float v = acc[r] * s + bb;
        v = v > 0.f ? v : 0.f;   // ReLU
        dstFrag[(((size_t)rowTile * KTD + kTileD) * 32 + (half * 16 + mBase + r)) * 16 + e] =
            (_Float16)v;
      }
    }
  }
}

template <int DPREV, int KP1, int C0, int C1, int C2>
__global__ void __launch_bounds__(256)
edgeconv_kernel(const _Float16* __restrict__ fts16_in, const int* __restrict__ idx,
                const _Float16* __restrict__ w1, const _Float16* __restrict__ w2,
                const _Float16* __restrict__ w3,
                const float* __restrict__ s1, const float* __restrict__ b1,
                const float* __restrict__ s2, const float* __restrict__ b2,
                const float* __restrict__ s3, const float* __restrict__ b3,
                const float* __restrict__ skip32,
                float* __restrict__ fts32_out, _Float16* __restrict__ fts16_out) {
  extern __shared__ char smem[];
  constexpr int MAX0 = (KP1 > C1 ? KP1 : C1);
  constexpr int S0 = 8 * (MAX0 / 32) * 512;   // halves
  _Float16* buf0 = (_Float16*)smem;           // E, then H2
  _Float16* buf1 = buf0 + S0;                 // H1, then H3
  __shared__ int nidx[8][KNN];

  const int tid = threadIdx.x, lane = tid & 31, wave = tid >> 5;
  const int grp = blockIdx.x;                 // 2048 groups of 8 points
  const int b   = grp / (NP / 8);
  const int p0  = (grp % (NP / 8)) * 8;

  if (tid < 8 * KNN)
    nidx[tid >> 4][tid & 15] = idx[((size_t)(b * NP + p0 + (tid >> 4))) * KNN + (tid & 15)];
  __syncthreads();

  // -------- gather edge features [x_i, x_j - x_i] into A-frag layout ------
  constexpr int KT1 = KP1 / 32;
  if constexpr (DPREV % 8 == 0) {
    // vectorized: 8-channel chunks -> one/two global b128 loads + one ds b128
    constexpr int CH = KP1 / 8;               // chunks per row
    for (int t = tid; t < 128 * CH; t += 256) {
      int r = t / CH, cc = t % CH;
      int c0 = cc * 8;
      int p = r >> 4, slot = r & 15;
      int gi = b * NP + p0 + p;
      int gj = b * NP + nidx[p][slot];
      v8h val;
      if (c0 + 8 <= DPREV) {
        val = *(const v8h*)(fts16_in + (size_t)gi * DPREV + c0);
      } else if (c0 < 2 * DPREV) {
        int cb = c0 - DPREV;
        const v8h xi = *(const v8h*)(fts16_in + (size_t)gi * DPREV + cb);
        const v8h xj = *(const v8h*)(fts16_in + (size_t)gj * DPREV + cb);
        val = xj - xi;                         // packed f16
      } else {
#pragma unroll
        for (int e = 0; e < 8; ++e) val[e] = (_Float16)0.f;
      }
      int kTile = c0 >> 5, kk = c0 & 31;
      int half = (kk & 15) >> 3;
      int e0   = ((kk >> 4) << 3);             // 8-aligned chunk -> contiguous e
      *(v8h*)(buf0 + (((size_t)p * KT1 + kTile) * 32 + (half * 16 + slot)) * 16 + e0) = val;
    }
  } else {
    for (int t = tid; t < 128 * KP1; t += 256) {
      int r = t / KP1, c = t % KP1;
      int p = r >> 4, slot = r & 15;
      int gi = b * NP + p0 + p;
      int gj = b * NP + nidx[p][slot];
      _Float16 val;
      if (c < DPREV) {
        val = fts16_in[(size_t)gi * DPREV + c];
      } else if (c < 2 * DPREV) {
        int cc = c - DPREV;
        val = (_Float16)((float)fts16_in[(size_t)gj * DPREV + cc] -
                         (float)fts16_in[(size_t)gi * DPREV + cc]);
      } else {
        val = (_Float16)0.f;
      }
      int kTile = c >> 5, kk = c & 31;
      int half = (kk & 15) >> 3;
      int e    = ((kk >> 4) << 3) | (kk & 7);
      buf0[(((size_t)p * KT1 + kTile) * 32 + (half * 16 + slot)) * 16 + e] = val;
    }
  }
  __syncthreads();

  wg_gemm<KP1, C0>(buf0, buf1, w1, s1, b1, lane, wave);
  __syncthreads();
  wg_gemm<C0, C1>(buf1, buf0, w2, s2, b2, lane, wave);
  __syncthreads();
  wg_gemm<C1, C2>(buf0, buf1, w3, s3, b3, lane, wave);
  __syncthreads();

  // -------- max over 16 edges per point + skip + ReLU ---------------------
  constexpr int KT3 = C2 / 32;
  for (int t = tid; t < 8 * C2; t += 256) {
    int p = t / C2, c = t % C2;
    int kTile = c >> 5, kk = c & 31;
    int half = (kk & 15) >> 3;
    int e    = ((kk >> 4) << 3) | (kk & 7);
    const _Float16* base = buf1 + (((size_t)p * KT3 + kTile) * 32) * 16 + e;
    float mx = -1e30f;
#pragma unroll
    for (int m = 0; m < 16; ++m) {
      float v = (float)base[(half * 16 + m) * 16];
      mx = mx > v ? mx : v;
    }
    size_t row = (size_t)(b * NP + p0 + p);
    float o = mx + skip32[row * C2 + c];
    o = o > 0.f ? o : 0.f;
    fts32_out[row * C2 + c] = o;
    fts16_out[row * C2 + c] = (_Float16)o;
  }
}

// ============================ head ========================================
__global__ void pool_kernel(const float* __restrict__ fts, float* __restrict__ g) {
  int b = blockIdx.x, c = threadIdx.x;   // 256 channels
  float s = 0.f;
  for (int n = 0; n < NP; ++n) s += fts[((size_t)(b * NP + n)) * 256 + c];
  g[b * 256 + c] = s * (1.0f / NP);
}

__global__ void fc_kernel(const float* __restrict__ g, const float* __restrict__ W,
                          const float* __restrict__ bias, float* __restrict__ h) {
  int b = blockIdx.x, c = threadIdx.x;
  float s = bias[c];
  for (int k = 0; k < 256; ++k) s += g[b * 256 + k] * W[k * 256 + c];
  h[b * 256 + c] = s > 0.f ? s : 0.f;
}

__global__ void out_kernel(const float* __restrict__ h, const float* __restrict__ W,
                           const float* __restrict__ bias, float* __restrict__ out) {
  int b = threadIdx.x;
  if (b >= BJ) return;
  float s = bias[0];
  for (int k = 0; k < 256; ++k) s += h[b * 256 + k] * W[k];
  out[b] = 1.0f / (1.0f + expf(-s));
}

// ============================ launcher ====================================
// Leaf order assumes JAX sorted-key tree flattening of {"params","pos","x"}:
//  convs[l]: Ws[0..2], Wskip, bn_skip{beta,gamma,mean,var}, bns[0..2]{b,g,m,v}
//  then fc_W, fc_b, input_bn{b,g,m,v}, out_W, out_b, pos, x.
#define IN_POS 68
#define IN_X   69

static inline unsigned cdiv(unsigned a, unsigned b) { return (a + b - 1) / b; }

extern "C" void kernel_launch(void* const* d_in, const int* in_sizes, int n_in,
                              void* d_out, int out_size, void* d_ws, size_t ws_size,
                              hipStream_t stream) {
  (void)in_sizes; (void)n_in; (void)out_size; (void)ws_size;
  char* ws = (char*)d_ws;

  // workspace carve-out (bytes)
  float*     fts32A = (float*)    (ws + (size_t)0);
  float*     fts32B = (float*)    (ws + ((size_t)16 << 20));
  _Float16*  fts16A = (_Float16*) (ws + ((size_t)32 << 20));
  _Float16*  fts16B = (_Float16*) (ws + ((size_t)40 << 20));
  float*     skip   = (float*)    (ws + ((size_t)48 << 20));
  int*       knnidx = (int*)      (ws + ((size_t)64 << 20));
  char*      wfbase =              ws + ((size_t)65 << 20);
  char*      sbbase =              ws + ((size_t)66 << 20);
  float*     gpool  = (float*)    (ws + ((size_t)67 << 20));
  float*     fcout  = (float*)    (ws + ((size_t)67 << 20) + (1 << 17));

  // weight-fragment slots
  _Float16* wfC1[4] = { (_Float16*)(wfbase + 0),      (_Float16*)(wfbase + 4096),
                        (_Float16*)(wfbase + 12288),  (_Float16*)(wfbase + 20480) };
  _Float16* wfC2[4] = { (_Float16*)(wfbase + 24576),  (_Float16*)(wfbase + 57344),
                        (_Float16*)(wfbase + 90112),  (_Float16*)(wfbase + 122880) };
  _Float16* wfC3[4] = { (_Float16*)(wfbase + 139264), (_Float16*)(wfbase + 270336),
                        (_Float16*)(wfbase + 401408), (_Float16*)(wfbase + 532480) };

  auto sbS = [&](int s) { return (float*)(sbbase + (size_t)s * 2048); };
  auto sbB = [&](int s) { return (float*)(sbbase + (size_t)s * 2048 + 1024); };

  // ---- input BN -> fts A (stride 7) ----
  input_bn_kernel<<<cdiv(BJ * NP * 7, 256), 256, 0, stream>>>(
      (const float*)d_in[IN_X], (const float*)d_in[62], (const float*)d_in[63],
      (const float*)d_in[64], (const float*)d_in[65], fts32A, fts16A);

  // ---- BN fold: sets l*4 + {bns0,bns1,bns2,bn_skip} ----
  const int convBase[3] = { 0, 20, 40 };
  const int convC[3]    = { 64, 128, 256 };
  for (int l = 0; l < 3; ++l) {
    int base = convBase[l], C = convC[l];
    for (int j = 0; j < 3; ++j) {  // bns[j] leaves at base+8+4*j
      int p = base + 8 + 4 * j;
      bnfold_kernel<<<cdiv(C, 256), 256, 0, stream>>>(
          (const float*)d_in[p], (const float*)d_in[p + 1],
          (const float*)d_in[p + 2], (const float*)d_in[p + 3],
          sbS(l * 4 + j), sbB(l * 4 + j), C);
    }
    int p = base + 4;              // bn_skip leaves
    bnfold_kernel<<<cdiv(C, 256), 256, 0, stream>>>(
        (const float*)d_in[p], (const float*)d_in[p + 1],
        (const float*)d_in[p + 2], (const float*)d_in[p + 3],
        sbS(l * 4 + 3), sbB(l * 4 + 3), C);
  }

  // ---- weight swizzle (K, Kpad, Nc) ----
  struct WSpec { int leaf; _Float16* dst; int K, Kpad, Nc; };
  WSpec specs[12] = {
    { 0, wfC1[0], 14, 32, 64 }, { 1, wfC1[1], 64, 64, 64 },
    { 2, wfC1[2], 64, 64, 64 }, { 3, wfC1[3],  7, 32, 64 },
    {20, wfC2[0],128,128,128 }, {21, wfC2[1],128,128,128 },
    {22, wfC2[2],128,128,128 }, {23, wfC2[3], 64, 64,128 },
    {40, wfC3[0],256,256,256 }, {41, wfC3[1],256,256,256 },
    {42, wfC3[2],256,256,256 }, {43, wfC3[3],128,128,256 },
  };
  for (int i = 0; i < 12; ++i) {
    WSpec& w = specs[i];
    w2frag_kernel<<<cdiv(w.Kpad * w.Nc, 256), 256, 0, stream>>>(
        (const float*)d_in[w.leaf], w.dst, w.K, w.Kpad, w.Nc);
  }

  const size_t knnShm = (128 * 132 + 128 * 33 + 128) * sizeof(float);

  // ================= conv1: 7 -> 64, kNN on pos (d=2) =================
  knn_kernel<<<BJ, 128, knnShm, stream>>>((const float*)d_in[IN_POS], 2, knnidx);
  skip_gemm_kernel<7, 32, 64><<<(1024 * (64 / 16)) / 8, 256, 0, stream>>>(
      fts16A, wfC1[3], sbS(3), sbB(3), skip);
  edgeconv_kernel<7, 32, 64, 64, 64><<<BJ * NP / 8, 256, 32768, stream>>>(
      fts16A, knnidx, wfC1[0], wfC1[1], wfC1[2],
      sbS(0), sbB(0), sbS(1), sbB(1), sbS(2), sbB(2),
      skip, fts32B, fts16B);

  // ================= conv2: 64 -> 128, kNN in feature space ===========
  knn_kernel<<<BJ, 128, knnShm, stream>>>(fts32B, 64, knnidx);
  skip_gemm_kernel<64, 64, 128><<<(1024 * (128 / 16)) / 8, 256, 0, stream>>>(
      fts16B, wfC2[3], sbS(7), sbB(7), skip);
  edgeconv_kernel<64, 128, 128, 128, 128><<<BJ * NP / 8, 256, 65536, stream>>>(
      fts16B, knnidx, wfC2[0], wfC2[1], wfC2[2],
      sbS(4), sbB(4), sbS(5), sbB(5), sbS(6), sbB(6),
      skip, fts32A, fts16A);

  // ================= conv3: 128 -> 256 ================================
  knn_kernel<<<BJ, 128, knnShm, stream>>>(fts32A, 128, knnidx);
  skip_gemm_kernel<128, 128, 256><<<(1024 * (256 / 16)) / 8, 256, 0, stream>>>(
      fts16A, wfC3[3], sbS(11), sbB(11), skip);
  edgeconv_kernel<128, 256, 256, 256, 256><<<BJ * NP / 8, 256, 131072, stream>>>(
      fts16A, knnidx, wfC3[0], wfC3[1], wfC3[2],
      sbS(8), sbB(8), sbS(9), sbB(9), sbS(10), sbB(10),
      skip, fts32B, fts16B);

  // ================= head =============================================
  pool_kernel<<<BJ, 256, 0, stream>>>(fts32B, gpool);
  fc_kernel<<<BJ, 256, 0, stream>>>(gpool, (const float*)d_in[60],
                                    (const float*)d_in[61], fcout);
  out_kernel<<<1, 128, 0, stream>>>(fcout, (const float*)d_in[66],
                                    (const float*)d_in[67], (float*)d_out);
}
